// SoftDecisionTree_43525198578189
// MI455X (gfx1250) — matrix operand order repaired and checked by reading, not compiled, not run on hardware
//
#include <hip/hip_runtime.h>
#include <hip/hip_bf16.h>
#include <math.h>

// ---------------------------------------------------------------------------
// Soft decision tree on MI455X (gfx1250), wave32 + WMMA bf16.
//  logits(8192x63) = x(8192x4096) @ W.T  -> sigmoid -> depth-6 tree fold -> sum
// GEMM via v_wmma_f32_16x16x32_bf16; x converted f32->bf16 on the fly with a
// 3-op round-to-nearest pack (2x v_add_u32 + v_perm_b32). Non-temporal x loads
// protect the L2-resident W fragment buffer. Double-buffered LDS A-stage:
// 1 barrier/iter. Memory-bound target: ~128MB @ 23.3 TB/s ≈ 5.8 us.
// ---------------------------------------------------------------------------

typedef __attribute__((ext_vector_type(16))) __bf16 v16bf;
typedef __attribute__((ext_vector_type(8)))  float  v8f;
typedef __attribute__((ext_vector_type(4)))  float  v4f;

#define T_ROWS   8192
#define IN_SZ    4096
#define N_INT    63
#define N_LEAF   64
#define N_CLS    10
#define BM       128            // rows per workgroup
#define KC       (IN_SZ / 32)   // 128 K-chunks of 32
#define LDS_STRIDE 68           // u16 per LDS A row: 64 data + 4 pad (8B-aligned)
#define A_BUF_BYTES (BM * LDS_STRIDE * 2)   // 17408 B per buffer

// f32 pair -> packed bf16, round-to-nearest: 2x v_add_u32 + 1x v_perm_b32.
// perm source is {s0 = bytes 4..7, s1 = bytes 0..3}; select hi16 of each.
__device__ __forceinline__ unsigned pack_bf16(float a, float b) {
    union { float f; unsigned u; } ua, ub;
    ua.f = a; ub.f = b;
    return __builtin_amdgcn_perm(ub.u + 0x8000u, ua.u + 0x8000u, 0x07060302u);
}

__global__ void sdt_zero(float* out) { out[0] = 0.0f; }

// leaf_r[j] = softmax(leaf_dist[j,:]) . class_reward
__global__ void sdt_leaf(const float* __restrict__ leaf_dist,
                         const float* __restrict__ cr,
                         float* __restrict__ leafR) {
    int j = threadIdx.x;
    if (j < N_LEAF) {
        float m = -1e30f;
        for (int c = 0; c < N_CLS; ++c) m = fmaxf(m, leaf_dist[j * N_CLS + c]);
        float s = 0.f, acc = 0.f;
        for (int c = 0; c < N_CLS; ++c) {
            float e = __expf(leaf_dist[j * N_CLS + c] - m);
            s += e; acc += e * cr[c];
        }
        leafR[j] = acc / s;
    }
}

// Repack W (63x4096 f32, padded to 64 rows of zeros) into bf16 B-fragment
// order for V_WMMA_F32_16X16X32_BF16 (B is 32x16 K-major per chunk):
//   lane l: n = ntile*16 + (l&15), half = l>>4 ; VGPR i holds K = c*32 + half*16 + 2i (+1)
// flat u32 index: ((t*KC + c)*32 + lane)*8 + i        -> 131072 u32 = 512 KB
__global__ void sdt_wfrag(const float* __restrict__ W, unsigned* __restrict__ wsB) {
    int gid  = blockIdx.x * blockDim.x + threadIdx.x;
    int i    = gid & 7;
    int lane = (gid >> 3) & 31;
    int c    = (gid >> 8) & (KC - 1);
    int t    = gid >> 15;
    int n    = t * 16 + (lane & 15);
    int hh   = lane >> 4;
    int k    = c * 32 + hh * 16 + 2 * i;
    float a  = (n < N_INT) ? W[(size_t)n * IN_SZ + k]     : 0.f;
    float bv = (n < N_INT) ? W[(size_t)n * IN_SZ + k + 1] : 0.f;
    wsB[gid] = pack_bf16(a, bv);
}

__global__ __launch_bounds__(256) void sdt_main(
        const float* __restrict__ x,
        const float* __restrict__ b,
        const float* __restrict__ beta,
        const unsigned* __restrict__ wsB,
        const float* __restrict__ leafR,
        float* __restrict__ out) {
    // One raw arena, time-multiplexed:
    //  during K-loop : two bf16 A buffers   [0, 2*17408)
    //  after K-loop  : f32 P tile (128x64)  [0, 32768)
    __shared__ __align__(16) unsigned char lds[2 * A_BUF_BYTES];
    __shared__ float red[256];

    const int tid  = threadIdx.x;
    const int wave = tid >> 5;
    const int lane = tid & 31;
    const int hh   = lane >> 4;   // lane half (0/1)
    const int l16  = lane & 15;
    const long m0  = (long)blockIdx.x * BM;

    v8f acc[4] = {};   // 4 N-tiles of 16x16 f32

    for (int kb = 0; kb < IN_SZ; kb += 64) {
        unsigned short* ldsA =
            (unsigned short*)(lds + ((kb >> 6) & 1) * A_BUF_BYTES);

        // ---- stage 128x64 f32 -> bf16 into LDS (coalesced, non-temporal) ----
        #pragma unroll
        for (int it = 0; it < 8; ++it) {
            int f   = it * 256 + tid;       // 2048 float4s total
            int row = f >> 4;
            int c4  = f & 15;
            const v4f v = __builtin_nontemporal_load(
                (const v4f*)(x + (m0 + row) * IN_SZ + kb) + c4);
            uint2 pk;
            pk.x = pack_bf16(v.x, v.y);
            pk.y = pack_bf16(v.z, v.w);
            *(uint2*)(ldsA + row * LDS_STRIDE + c4 * 4) = pk;
        }
        __syncthreads();   // single barrier per iter (double-buffered A)

        // ---- two K=32 WMMA steps ----
        #pragma unroll
        for (int sub = 0; sub < 2; ++sub) {
            const int c = (kb >> 5) + sub;  // global K-chunk index
            // A fragment (16x32 bf16): lane holds row M=l16;
            // VGPR i: K = half*16 + 2i  (+ sub*32 within tile)
            union { uint2 q[4]; v16bf v; } af;
            const int mrow = wave * 16 + l16;
            #pragma unroll
            for (int g = 0; g < 2; ++g) {
                int koff = sub * 32 + hh * 16 + g * 8;   // u16 units
                af.q[g * 2 + 0] = *(const uint2*)(ldsA + mrow * LDS_STRIDE + koff);
                af.q[g * 2 + 1] = *(const uint2*)(ldsA + mrow * LDS_STRIDE + koff + 4);
            }
            #pragma unroll
            for (int t = 0; t < 4; ++t) {
                union { uint4 q[2]; v16bf v; } bfrag;
                const unsigned* p = wsB + (((size_t)(t * KC + c)) * 32 + lane) * 8;
                bfrag.q[0] = *(const uint4*)(p);
                bfrag.q[1] = *(const uint4*)(p + 4);
                acc[t] = __builtin_amdgcn_wmma_f32_16x16x32_bf16(
                    false, af.v, false, bfrag.v, (short)0, acc[t], false, false);
            }
        }
    }
    __syncthreads();       // retire all A-buffer reads before P reuses arena

    // ---- epilogue: bias + sigmoid, scatter P into the arena ----
    // C layout: vgpr r, lane l -> M = r + 8*(l>>4), N = l&15
    float* ldsP = (float*)lds;   // 128 x 64 f32
    #pragma unroll
    for (int t = 0; t < 4; ++t) {
        const int col = t * 16 + l16;
        const float bb = (col < N_INT) ? b[col]    : 0.f;
        const float bt = (col < N_INT) ? beta[col] : 0.f;
        #pragma unroll
        for (int r = 0; r < 8; ++r) {
            const int row = wave * 16 + hh * 8 + r;
            float logit = acc[t][r] + bb;
            float pval  = 1.f / (1.f + __expf(-bt * logit));
            ldsP[row * 64 + col] = (col < N_INT) ? pval : 0.f;
        }
    }
    __syncthreads();

    // ---- depth-6 tree fold per row: R[node] = P*R[2i] + (1-P)*R[2i+1] ----
    float val = 0.f;
    if (tid < BM) {
        const float* Prow = ldsP + tid * 64;
        float R[32];
        #pragma unroll
        for (int i = 0; i < 32; ++i) {
            float p = Prow[31 + i];
            R[i] = p * leafR[2 * i] + (1.f - p) * leafR[2 * i + 1];
        }
        #pragma unroll
        for (int n = 16; n >= 1; n >>= 1) {
            #pragma unroll
            for (int i = 0; i < 16; ++i) {
                if (i < n) {
                    float p = Prow[(n - 1) + i];
                    R[i] = p * R[2 * i] + (1.f - p) * R[2 * i + 1];
                }
            }
        }
        val = R[0];
    }

    // ---- block reduction + global atomic ----
    red[tid] = val;
    __syncthreads();
    #pragma unroll
    for (int s = 128; s > 0; s >>= 1) {
        if (tid < s) red[tid] += red[tid + s];
        __syncthreads();
    }
    if (tid == 0) atomicAdd(out, red[0]);
}

extern "C" void kernel_launch(void* const* d_in, const int* in_sizes, int n_in,
                              void* d_out, int out_size, void* d_ws, size_t ws_size,
                              hipStream_t stream) {
    const float* x    = (const float*)d_in[0];
    const float* W    = (const float*)d_in[1];
    const float* b    = (const float*)d_in[2];
    const float* beta = (const float*)d_in[3];
    const float* ld   = (const float*)d_in[4];
    const float* cr   = (const float*)d_in[5];
    float* out = (float*)d_out;

    unsigned* wsB  = (unsigned*)d_ws;                       // 512 KB fragments
    float*    lfR  = (float*)((char*)d_ws + (size_t)4 * KC * 32 * 8 * sizeof(unsigned));

    sdt_zero <<<1,   1,   0, stream>>>(out);
    sdt_leaf <<<1,   64,  0, stream>>>(ld, cr, lfR);
    sdt_wfrag<<<512, 256, 0, stream>>>(W, wsB);
    sdt_main <<<64,  256, 0, stream>>>(x, b, beta, wsB, lfR, out);
}
